// GAT_60078002536565
// MI455X (gfx1250) — matrix operand order repaired and checked by reading, not compiled, not run on hardware
//
#include <hip/hip_runtime.h>

#define NN 50000
#define NE 400000
#define INF 256
#define HID 64
#define NH 8
#define NC 40
#define NEG_SLOPE 0.2f

typedef __attribute__((ext_vector_type(16))) _Float16 v16h;
typedef __attribute__((ext_vector_type(8)))  _Float16 v8h;
typedef __attribute__((ext_vector_type(8)))  float    v8f;

// ---------------- conversion / utility kernels ----------------

__global__ void k_f32_to_f16(const float* __restrict__ s, _Float16* __restrict__ d, long n) {
  long i = (long)blockIdx.x * blockDim.x + threadIdx.x;
  if (i < n) d[i] = (_Float16)s[i];
}

// W [K][Nout] f32 (row-major) -> BT [Npad][K] f16, zero-padding cols >= Nout
__global__ void k_w_to_bt(const float* __restrict__ W, _Float16* __restrict__ BT,
                          int K, int Nout, int Npad) {
  int i = blockIdx.x * blockDim.x + threadIdx.x;
  int total = Npad * K;
  if (i >= total) return;
  int col = i / K;
  int k = i - col * K;
  BT[i] = (col < Nout) ? (_Float16)W[(long)k * Nout + col] : (_Float16)0.f;
}

__global__ void k_zero32(unsigned* __restrict__ p, long n) {
  long i = (long)blockIdx.x * blockDim.x + threadIdx.x;
  if (i < n) p[i] = 0u;
}

// ---------------- WMMA GEMM: C[M][Nout] = A[M][K] * BT[Npad][K]^T ----------------
// One wave per 16x64 output block (4 N-tiles register-blocked): each A fragment
// feeds 4 v_wmma_f32_16x16x32_f16 ops. A,BT f16; C f32. Npad must be a
// multiple of 64; M a multiple of 16 (50000 = 3125*16).
// Fragment layouts used by this kernel (wave32; A/B k order is contiguous, not ISA table order):
//   A 16x32 f16: lane&15 = M row; half j -> K=(lane>>4)*16+j (contiguous, matches B)
//   B 32x16 f16: lane&15 = N col; half j -> K=(lane>>4)*16+j (BT pre-transposed => contiguous)
//   C 16x16 f32: lane&15 = N col; VGPR i -> M=(lane>>4)*8+i
__global__ void __launch_bounds__(256)
k_wmma_gemm(const _Float16* __restrict__ A, const _Float16* __restrict__ BT,
            float* __restrict__ C, int M, int K, int Npad, int Nout) {
  const int lane  = threadIdx.x & 31;
  const int wave  = threadIdx.x >> 5;
  const int nBlk  = Npad >> 6;               // 64-wide N blocks
  const int mTiles = M >> 4;
  int tile = blockIdx.x * 8 + wave;
  if (tile >= mTiles * nBlk) return;         // wave-uniform exit: EXEC all-1 for WMMA
  const int mt = tile / nBlk;
  const int nb = tile - mt * nBlk;
  const int hl  = lane >> 4;                 // 0 | 1
  const int l15 = lane & 15;
  const long arow  = (long)(mt * 16 + l15) * K;
  const long brow  = (long)(nb * 64 + l15) * K;
  const long bstep = (long)16 * K;           // next 16-col group of BT
  const int kA = hl * 16;
  const int kB = hl * 16;

  v8f acc[4] = {v8f{}, v8f{}, v8f{}, v8f{}};
  for (int k0 = 0; k0 < K; k0 += 32) {
    v8h a0 = *(const v8h*)(A + arow + k0 + kA);
    v8h a1 = *(const v8h*)(A + arow + k0 + kA + 8);
    v16h a;
#pragma unroll
    for (int i = 0; i < 8; ++i) { a[i] = a0[i]; a[i + 8] = a1[i]; }
#pragma unroll
    for (int t = 0; t < 4; ++t) {
      v16h b = *(const v16h*)(BT + brow + t * bstep + k0 + kB);
      acc[t] = __builtin_amdgcn_wmma_f32_16x16x32_f16(false, a, false, b,
                                                      (short)0, acc[t], false, false);
    }
  }

#pragma unroll
  for (int t = 0; t < 4; ++t) {
    int colOut = nb * 64 + t * 16 + l15;
    if (colOut < Nout) {
#pragma unroll
      for (int i = 0; i < 8; ++i) {
        int row = mt * 16 + hl * 8 + i;
        C[(long)row * Nout + colOut] = acc[t][i];
      }
    }
  }
}

// ---------------- attention scalar products: el/er [N*H] ----------------
__global__ void k_attn_dots(const float* __restrict__ feat, const float* __restrict__ al,
                            const float* __restrict__ ar, float* __restrict__ el,
                            float* __restrict__ er, int H, int D) {
  int i = blockIdx.x * blockDim.x + threadIdx.x;   // i = n*H + h
  if (i >= NN * H) return;
  int h = i % H;
  const float* f = feat + (long)i * D;             // feat[n][h][d] contiguous
  const float* a = al + h * D;
  const float* b = ar + h * D;
  float sl = 0.f, sr = 0.f;
  for (int d = 0; d < D; ++d) { float v = f[d]; sl += v * a[d]; sr += v * b[d]; }
  el[i] = sl;
  er[i] = sr;
}

// ---------------- edge softmax ----------------
__device__ __forceinline__ unsigned enc_f32(float f) {
  unsigned u = __float_as_uint(f);
  return (u & 0x80000000u) ? ~u : (u | 0x80000000u);
}
__device__ __forceinline__ float dec_f32(unsigned u) {
  return (u & 0x80000000u) ? __uint_as_float(u ^ 0x80000000u) : __uint_as_float(~u);
}

__global__ void k_edge_max(const int* __restrict__ src, const int* __restrict__ dst,
                           const float* __restrict__ el, const float* __restrict__ er,
                           float* __restrict__ ee, unsigned* __restrict__ mx, int H) {
  long i = (long)blockIdx.x * blockDim.x + threadIdx.x;
  if (i >= (long)NE * H) return;
  int e = (int)(i / H);
  int h = (int)(i - (long)e * H);
  float v = el[(long)src[e] * H + h] + er[(long)dst[e] * H + h];
  v = (v > 0.f) ? v : v * NEG_SLOPE;               // leaky relu
  ee[i] = v;
  atomicMax(mx + (long)dst[e] * H + h, enc_f32(v));
}

__global__ void k_edge_exp(const int* __restrict__ dst, const unsigned* __restrict__ mx,
                           float* __restrict__ ee, float* __restrict__ denom, int H) {
  long i = (long)blockIdx.x * blockDim.x + threadIdx.x;
  if (i >= (long)NE * H) return;
  int e = (int)(i / H);
  int h = (int)(i - (long)e * H);
  float m = dec_f32(mx[(long)dst[e] * H + h]);
  float x = __expf(ee[i] - m);
  ee[i] = x;
  atomicAdd(denom + (long)dst[e] * H + h, x);
}

// one block per edge; scatter alpha*feat[src] into out[dst]
__global__ void k_edge_agg(const int* __restrict__ src, const int* __restrict__ dst,
                           const float* __restrict__ ee, const float* __restrict__ denom,
                           const float* __restrict__ feat, float* __restrict__ out,
                           int H, int D) {
  int e = blockIdx.x;
  int s = src[e], t = dst[e];
  int HD = H * D;
  for (int d = threadIdx.x; d < HD; d += blockDim.x) {
    int h = d / D;
    float alpha = ee[(long)e * H + h] / denom[(long)t * H + h];
    atomicAdd(out + (long)t * HD + d, alpha * feat[(long)s * HD + d]);
  }
}

// ---------------- host orchestration ----------------

struct Ws {
  float* feat; float* agg; _Float16* A16; _Float16* BT16;
  float* el; float* er; unsigned* mx; float* denom; float* ee;
};

static void run_layer(const float* hin, int K, const float* W,
                      const float* al, const float* ar,
                      int H, int D, int Nout, int Npad,
                      float* out, const int* src, const int* dst,
                      const Ws& w, hipStream_t stream) {
  const long nA = (long)NN * K;
  k_f32_to_f16<<<(unsigned)((nA + 255) / 256), 256, 0, stream>>>(hin, w.A16, nA);

  const int nB = Npad * K;
  k_w_to_bt<<<(nB + 255) / 256, 256, 0, stream>>>(W, w.BT16, K, Nout, Npad);

  const int waves = (NN / 16) * (Npad / 64);   // one wave per 16x64 block
  k_wmma_gemm<<<(waves + 7) / 8, 256, 0, stream>>>(w.A16, w.BT16, w.feat, NN, K, Npad, Nout);

  const int nh = NN * H;
  k_attn_dots<<<(nh + 255) / 256, 256, 0, stream>>>(w.feat, al, ar, w.el, w.er, H, D);

  k_zero32<<<(nh + 255) / 256, 256, 0, stream>>>(w.mx, nh);          // enc(-inf) sentinel
  k_zero32<<<(nh + 255) / 256, 256, 0, stream>>>((unsigned*)w.denom, nh);

  const long eh = (long)NE * H;
  k_edge_max<<<(unsigned)((eh + 255) / 256), 256, 0, stream>>>(src, dst, w.el, w.er, w.ee, w.mx, H);
  k_edge_exp<<<(unsigned)((eh + 255) / 256), 256, 0, stream>>>(dst, w.mx, w.ee, w.denom, H);

  const long nout = (long)NN * H * D;
  k_zero32<<<(unsigned)((nout + 255) / 256), 256, 0, stream>>>((unsigned*)out, nout);
  int blk = (H * D >= 128) ? 128 : 64;
  k_edge_agg<<<NE, blk, 0, stream>>>(src, dst, w.ee, w.denom, w.feat, out, H, D);
}

extern "C" void kernel_launch(void* const* d_in, const int* in_sizes, int n_in,
                              void* d_out, int out_size, void* d_ws, size_t ws_size,
                              hipStream_t stream) {
  const float* features = (const float*)d_in[0];
  const int*   src      = (const int*)d_in[1];
  const int*   dst      = (const int*)d_in[2];
  const float* W1 = (const float*)d_in[3];
  const float* al1 = (const float*)d_in[4];
  const float* ar1 = (const float*)d_in[5];
  const float* W2 = (const float*)d_in[6];
  const float* al2 = (const float*)d_in[7];
  const float* ar2 = (const float*)d_in[8];
  const float* W3 = (const float*)d_in[9];
  const float* al3 = (const float*)d_in[10];
  const float* ar3 = (const float*)d_in[11];

  char* base = (char*)d_ws;
  size_t off = 0;
  auto carve = [&](size_t bytes) -> void* {
    void* p = base + off;
    off += (bytes + 255) & ~(size_t)255;
    return p;
  };
  Ws w;
  w.feat  = (float*)carve((size_t)NN * 512 * 4);     // GEMM output (per-layer)
  w.agg   = (float*)carve((size_t)NN * 512 * 4);     // aggregated activations
  w.A16   = (_Float16*)carve((size_t)NN * 512 * 2);  // f16 activations
  w.BT16  = (_Float16*)carve((size_t)512 * 512 * 2); // f16 transposed weights (padded)
  w.el    = (float*)carve((size_t)NN * NH * 4);
  w.er    = (float*)carve((size_t)NN * NH * 4);
  w.mx    = (unsigned*)carve((size_t)NN * NH * 4);
  w.denom = (float*)carve((size_t)NN * NH * 4);
  w.ee    = (float*)carve((size_t)NE * NH * 4);

  // Layer 1: 256 -> 8x64 (Nout = Npad = 512)
  run_layer(features, INF, W1, al1, ar1, NH, HID, NH * HID, NH * HID,
            w.agg, src, dst, w, stream);
  // Layer 2: 512 -> 8x64 (reads w.agg via f16 convert before w.agg is re-zeroed)
  run_layer(w.agg, NH * HID, W2, al2, ar2, NH, HID, NH * HID, NH * HID,
            w.agg, src, dst, w, stream);
  // Layer 3: 512 -> 1x40, pad B cols to 64 (one 16x64 block per M tile), write to d_out
  run_layer(w.agg, NH * HID, W3, al3, ar3, 1, NC, NC, 64,
            (float*)d_out, src, dst, w, stream);
}
